// Attention_74732430950411
// MI455X (gfx1250) — matrix operand order repaired and checked by reading, not compiled, not run on hardware
//
#include <hip/hip_runtime.h>
#include <hip/hip_bf16.h>

typedef __attribute__((ext_vector_type(16))) _Float16 v16h;
typedef __attribute__((ext_vector_type(8)))  _Float16 v8h;
typedef __attribute__((ext_vector_type(8)))  float    v8f;

#define B_SZ   128
#define S_SZ   1024
#define VO_SZ  512
#define H_SZ   8
#define HD_SZ  64
#define HALF_D 256
#define WAVES  8

__device__ __forceinline__ float4 ldg4(const float* p) {
    return *reinterpret_cast<const float4*>(p);
}

// ---------------------------------------------------------------------------
// Kernel 1: per-batch prep.
//   q[b]         = x[b] @ Wq^T                      (512)
//   wcomb[b,d,h] = (1/8) * sum_j Wk[h*64+j, d] * q[b, h*64+j]   -> f16, [b][d][h]
// so that scores[b,h,s] = rope(keys[b,s]) . wcomb[b,:,h]
// ---------------------------------------------------------------------------
__global__ __launch_bounds__(256)
void attn_prep_kernel(const float* __restrict__ x,
                      const float* __restrict__ Wq,
                      const float* __restrict__ Wk,
                      _Float16* __restrict__ wcomb)
{
    __shared__ float xs[VO_SZ];
    __shared__ float qs[VO_SZ];
    const int b   = blockIdx.x;
    const int tid = threadIdx.x;

    xs[tid]       = x[(size_t)b * VO_SZ + tid];
    xs[tid + 256] = x[(size_t)b * VO_SZ + tid + 256];
    __syncthreads();

    for (int jj = 0; jj < 2; ++jj) {
        const int j = tid + jj * 256;
        const float* wr = Wq + (size_t)j * VO_SZ;
        float acc = 0.f;
        #pragma unroll 4
        for (int d = 0; d < VO_SZ; d += 4) {
            float4 w = ldg4(wr + d);
            acc += w.x * xs[d] + w.y * xs[d + 1] + w.z * xs[d + 2] + w.w * xs[d + 3];
        }
        qs[j] = acc;
    }
    __syncthreads();

    for (int dd = 0; dd < 2; ++dd) {
        const int d = tid + dd * 256;
        #pragma unroll
        for (int h = 0; h < H_SZ; ++h) {
            float acc = 0.f;
            #pragma unroll 8
            for (int j = 0; j < HD_SZ; ++j)
                acc += Wk[(size_t)(h * HD_SZ + j) * VO_SZ + d] * qs[h * HD_SZ + j];
            wcomb[((size_t)b * VO_SZ + d) * H_SZ + h] = (_Float16)(acc * 0.125f); // 1/sqrt(64)
        }
    }
}

// ---------------------------------------------------------------------------
// Kernel 2: one workgroup (8 wave32) per batch b.
//  0) stage wcomb into LDS pre-swizzled into WMMA B-operand layout (zero-padded N)
//  A) scores[s,h] = rope(keys[b,s]) . wcomb[b,:,h]        via v_wmma f16
//  B) softmax over s (per head, one wave per head); attn rows 8..15 zeroed
//  C) ctx[h,d]    = sum_s attn[h,s] * rope(states[b,s,d]) via v_wmma f16
//  D) out[b]      = (ctx @ Wv_h^T per head) @ Wo^T + x[b]  (VALU, tiny)
// ---------------------------------------------------------------------------
__global__ __launch_bounds__(256)
void attn_main_kernel(const float* __restrict__ x,
                      const float* __restrict__ keys,
                      const float* __restrict__ states,
                      const _Float16* __restrict__ wcomb,
                      const float* __restrict__ Wv,
                      const float* __restrict__ Wo,
                      float* __restrict__ out)
{
    __shared__ float    invf[HALF_D];            //  1 KB
    __shared__ float    scores[S_SZ * H_SZ];     // 32 KB
    __shared__ __align__(32) _Float16 attn16[16 * S_SZ];       // 32 KB (rows 8..15 = 0)
    __shared__ __align__(32) _Float16 bmat[16 * 32 * 16];      // 16 KB B-operands
    __shared__ float    ctx[H_SZ * VO_SZ];       // 16 KB
    __shared__ float    o1[VO_SZ];               //  2 KB

    const int b    = blockIdx.x;
    const int tid  = threadIdx.x;
    const int lane = tid & 31;
    const int wave = tid >> 5;
    const int hi   = lane >> 4;   // half-wave (selects K-halves in WMMA layouts)
    const int mn   = lane & 15;   // M (A rows) or N (B/D cols) within a tile

    // inv_freq[j] = 10000^(-j/256); angle(s,d) = s * inv_freq[d % 256]
    invf[tid & 255] = __expf(-(float)(tid & 255) * (9.210340371976184f / 256.0f));

    const _Float16* wcb   = wcomb + (size_t)b * VO_SZ * H_SZ;
    const size_t    kBase = (size_t)b * S_SZ * VO_SZ;

    // ---- Phase 0: build all 16 B-operand chunks (8 "lo" d-bases, 8 "hi") ----
    // Layout: bmat[(chunk*32 + lane)*16 + e] = B element K=(lane>>4)*8+e+(e>=8?8:0), N=lane&15.
    for (int i = 0; i < 2; ++i) {
        const int slot  = tid + i * 256;       // chunk*32 + lane, 0..511
        const int ck    = slot >> 5;           // 0..15
        const int l     = slot & 31;
        const int n     = l & 15;
        const int kh    = (l >> 4) * 8;
        const int dbase = (ck < 8) ? ck * 32 : (ck - 8) * 32 + HALF_D;
        _Float16 vals[16];
        if (n < H_SZ) {
            #pragma unroll
            for (int e = 0; e < 16; ++e)
                vals[e] = wcb[(dbase + kh + e + (e >= 8 ? 8 : 0)) * H_SZ + n];
        } else {
            #pragma unroll
            for (int e = 0; e < 16; ++e) vals[e] = (_Float16)0.f;
        }
        #pragma unroll
        for (int e = 0; e < 16; ++e) bmat[slot * 16 + e] = vals[e];
    }
    __syncthreads();

    // ---------------- Phase A: scores via WMMA ----------------
    for (int t = wave; t < S_SZ / 16; t += WAVES) {
        const int s = t * 16 + mn;                 // A-row this lane supplies
        const float* rowp = keys + kBase + (size_t)s * VO_SZ;
        v8f acc = {};
        #pragma unroll
        for (int kp = 0; kp < 8; ++kp) {           // d in [kp*32, kp*32+32) ∪ +256
            const int dlo = kp * 32;
            float xl[16], xh[16];
            {   // e = 0..7 : d = dlo + hi*8 + e
                const int d0 = dlo + hi * 8;
                float4 a0 = ldg4(rowp + d0);
                float4 a1 = ldg4(rowp + d0 + 4);
                float4 c0 = ldg4(rowp + d0 + HALF_D);
                float4 c1 = ldg4(rowp + d0 + HALF_D + 4);
                xl[0]=a0.x; xl[1]=a0.y; xl[2]=a0.z; xl[3]=a0.w;
                xl[4]=a1.x; xl[5]=a1.y; xl[6]=a1.z; xl[7]=a1.w;
                xh[0]=c0.x; xh[1]=c0.y; xh[2]=c0.z; xh[3]=c0.w;
                xh[4]=c1.x; xh[5]=c1.y; xh[6]=c1.z; xh[7]=c1.w;
                // e = 8..15 : d = dlo + 16 + hi*8 + (e-8)
                const int d1 = dlo + 16 + hi * 8;
                float4 a2 = ldg4(rowp + d1);
                float4 a3 = ldg4(rowp + d1 + 4);
                float4 c2 = ldg4(rowp + d1 + HALF_D);
                float4 c3 = ldg4(rowp + d1 + HALF_D + 4);
                xl[8]=a2.x;  xl[9]=a2.y;  xl[10]=a2.z; xl[11]=a2.w;
                xl[12]=a3.x; xl[13]=a3.y; xl[14]=a3.z; xl[15]=a3.w;
                xh[8]=c2.x;  xh[9]=c2.y;  xh[10]=c2.z; xh[11]=c2.w;
                xh[12]=c3.x; xh[13]=c3.y; xh[14]=c3.z; xh[15]=c3.w;
            }
            v16h alo, ahi;
            #pragma unroll
            for (int e = 0; e < 16; ++e) {
                const int kl = (e < 8) ? (hi * 8 + e) : (16 + hi * 8 + (e - 8));
                const int d  = dlo + kl;
                float sn, c;
                __sincosf((float)s * invf[d], &sn, &c);
                alo[e] = (_Float16)(xl[e] * c - xh[e] * sn);   // rope, d < 256
                ahi[e] = (_Float16)(xh[e] * c + xl[e] * sn);   // rope, d >= 256
            }
            const v16h blo = *reinterpret_cast<const v16h*>(&bmat[(kp * 32 + lane) * 16]);
            const v16h bhi = *reinterpret_cast<const v16h*>(&bmat[((8 + kp) * 32 + lane) * 16]);
            acc = __builtin_amdgcn_wmma_f32_16x16x32_f16(false, alo, false, blo,
                                                         (short)0, acc, false, false);
            acc = __builtin_amdgcn_wmma_f32_16x16x32_f16(false, ahi, false, bhi,
                                                         (short)0, acc, false, false);
        }
        if (mn < H_SZ) {   // D layout: lane holds N=mn, rows M = hi*8 + r
            #pragma unroll
            for (int r = 0; r < 8; ++r)
                scores[(t * 16 + hi * 8 + r) * H_SZ + mn] = acc[r];
        }
        if (t + WAVES < S_SZ / 16)
            __builtin_prefetch(keys + kBase + (size_t)((t + WAVES) * 16 + mn) * VO_SZ, 0, 1);
    }
    __syncthreads();

    // ---------------- Phase B: softmax (wave w -> head w) ----------------
    {
        const int h = wave;
        float mx = -1e30f;
        #pragma unroll 4
        for (int i = 0; i < S_SZ / 32; ++i)
            mx = fmaxf(mx, scores[(lane + i * 32) * H_SZ + h]);
        #pragma unroll
        for (int off = 16; off > 0; off >>= 1)
            mx = fmaxf(mx, __shfl_xor(mx, off, 32));
        float sum = 0.f;
        #pragma unroll 4
        for (int i = 0; i < S_SZ / 32; ++i) {
            const int s = lane + i * 32;
            const float p = __expf(scores[s * H_SZ + h] - mx);
            scores[s * H_SZ + h] = p;
            sum += p;
        }
        #pragma unroll
        for (int off = 16; off > 0; off >>= 1)
            sum += __shfl_xor(sum, off, 32);
        const float inv = 1.0f / sum;
        #pragma unroll 4
        for (int i = 0; i < S_SZ / 32; ++i) {
            const int s = lane + i * 32;
            attn16[h * S_SZ + s] = (_Float16)(scores[s * H_SZ + h] * inv);
        }
    }
    // zero attn rows 8..15 (vectorized 16B stores: 1024 v8h slots / 256 threads)
    {
        v8h z = {};
        v8h* dst = reinterpret_cast<v8h*>(attn16 + H_SZ * S_SZ);
        #pragma unroll
        for (int i = 0; i < 4; ++i)
            dst[tid * 4 + i] = z;
    }
    __syncthreads();

    // ---------------- Phase C: ctx = attn^T @ rope(states) via WMMA ----------------
    for (int pp = 0; pp < 2; ++pp) {
        const int p = wave + pp * WAVES;     // d-pair-tile 0..15 (d < 256, partner +256)
        const int d = p * 16 + mn;           // B column this lane supplies
        const float invfd = invf[d];
        const _Float16* arow = attn16 + mn * S_SZ;   // rows 8..15 are zero
        v8f accL = {}, accH = {};
        for (int sc = 0; sc < S_SZ / 32; ++sc) {
            const v8h alo = *reinterpret_cast<const v8h*>(&arow[sc * 32 + hi * 8]);
            const v8h ahi = *reinterpret_cast<const v8h*>(&arow[sc * 32 + 16 + hi * 8]);
            const v16h a = __builtin_shufflevector(alo, ahi,
                0, 1, 2, 3, 4, 5, 6, 7, 8, 9, 10, 11, 12, 13, 14, 15);
            v16h bl, bh;
            #pragma unroll
            for (int e = 0; e < 16; ++e) {   // B: roped states, K = s, N = d
                const int s = sc * 32 + hi * 8 + e + (e >= 8 ? 8 : 0);
                const float* rp = states + kBase + (size_t)s * VO_SZ;
                const float xlo = rp[d];
                const float xhi = rp[d + HALF_D];
                float sn, c;
                __sincosf((float)s * invfd, &sn, &c);
                bl[e] = (_Float16)(xlo * c - xhi * sn);
                bh[e] = (_Float16)(xhi * c + xlo * sn);
            }
            accL = __builtin_amdgcn_wmma_f32_16x16x32_f16(false, a, false, bl,
                                                          (short)0, accL, false, false);
            accH = __builtin_amdgcn_wmma_f32_16x16x32_f16(false, a, false, bh,
                                                          (short)0, accH, false, false);
        }
        if (hi == 0) {       // rows M = 0..7 (valid heads) live in lanes 0..15
            #pragma unroll
            for (int r = 0; r < 8; ++r) {
                ctx[r * VO_SZ + d]          = accL[r];
                ctx[r * VO_SZ + d + HALF_D] = accH[r];
            }
        }
    }
    __syncthreads();

    // ---------------- Phase D: head projection (Wv) + output projection (Wo) + residual ----
    for (int jj = 0; jj < 2; ++jj) {
        const int j = tid + jj * 256;
        const int h = j >> 6;
        const float* wr = Wv + (size_t)j * VO_SZ;
        const float* cr = ctx + h * VO_SZ;
        float acc = 0.f;
        #pragma unroll 4
        for (int d = 0; d < VO_SZ; d += 4) {
            float4 w = ldg4(wr + d);
            acc += w.x * cr[d] + w.y * cr[d + 1] + w.z * cr[d + 2] + w.w * cr[d + 3];
        }
        o1[j] = acc;
    }
    __syncthreads();
    for (int jj = 0; jj < 2; ++jj) {
        const int jo = tid + jj * 256;
        const float* wr = Wo + (size_t)jo * VO_SZ;
        float acc = x[(size_t)b * VO_SZ + jo];
        #pragma unroll 4
        for (int d = 0; d < VO_SZ; d += 4) {
            float4 w = ldg4(wr + d);
            acc += w.x * o1[d] + w.y * o1[d + 1] + w.z * o1[d + 2] + w.w * o1[d + 3];
        }
        out[(size_t)b * VO_SZ + jo] = acc;
    }
}

// ---------------------------------------------------------------------------
extern "C" void kernel_launch(void* const* d_in, const int* in_sizes, int n_in,
                              void* d_out, int out_size, void* d_ws, size_t ws_size,
                              hipStream_t stream)
{
    const float* x      = (const float*)d_in[0];
    const float* keys   = (const float*)d_in[1];
    const float* states = (const float*)d_in[2];
    const float* Wq     = (const float*)d_in[3];
    const float* Wk     = (const float*)d_in[4];
    const float* Wv     = (const float*)d_in[5];
    const float* Wo     = (const float*)d_in[6];

    _Float16* wcomb = (_Float16*)d_ws;   // 128 * 512 * 8 f16 = 1 MB scratch

    attn_prep_kernel<<<B_SZ, 256, 0, stream>>>(x, Wq, Wk, wcomb);
    attn_main_kernel<<<B_SZ, 256, 0, stream>>>(x, keys, states, wcomb, Wv, Wo,
                                               (float*)d_out);
}
